// MixtureOfExpertsES_49443663512011
// MI455X (gfx1250) — matrix operand order repaired and checked by reading, not compiled, not run on hardware
//
#include <hip/hip_runtime.h>

// Problem constants (match reference)
#define S_TOK 4096   // B*T
#define DMODEL 768
#define DFF   3072
#define NEXP  8

#define MTILE 32                 // tokens per workgroup tile
#define LDA (DMODEL + 8)         // bf16 elements, padded vs 64-bank LDS
#define LDH (DFF + 8)

#define NT1 (DFF / 16)           // 192 n-tiles, GEMM1
#define KS1 (DMODEL / 32)        // 24 k-steps,  GEMM1
#define NT2 (DMODEL / 16)        // 48 n-tiles,  GEMM2
#define KS2 (DFF / 32)           // 96 k-steps,  GEMM2
#define W1S_ELEMS ((size_t)NEXP * NT1 * KS1 * 512)   // 18,874,368 bf16
#define W2S_ELEMS ((size_t)NEXP * NT2 * KS2 * 512)

typedef __bf16 bf16;
typedef bf16  v16bf __attribute__((ext_vector_type(16)));
typedef bf16  v8bf  __attribute__((ext_vector_type(8)));
typedef bf16  v4bf  __attribute__((ext_vector_type(4)));
typedef float v8f   __attribute__((ext_vector_type(8)));

__device__ __forceinline__ bf16 f2bf(float f) {
  // round-to-nearest-even fp32 -> bf16 (off the hot GEMM loop)
  unsigned u = __builtin_bit_cast(unsigned, f);
  unsigned r = (u + 0x7FFFu + ((u >> 16) & 1u)) >> 16;
  unsigned short h = (unsigned short)r;
  return __builtin_bit_cast(bf16, h);
}

// ---------------------------------------------------------------- zero kernel
__global__ void moe_zero(float* __restrict__ out, int n, int* __restrict__ counts) {
  for (int i = blockIdx.x * blockDim.x + threadIdx.x; i < n; i += gridDim.x * blockDim.x)
    out[i] = 0.0f;
  if (blockIdx.x == 0 && threadIdx.x < NEXP) counts[threadIdx.x] = 0;
}

// --------------------------------------------- weight pre-swizzle fp32->bf16
// Dest layout (per matrix): frag[(e*NT + nt)*KS + ks][lane][i], 512 bf16/frag.
// Lane semantics per ISA 16-bit B layout: lane = half*16 + lr holds
// B[ks*32 + (i/8)*16 + half*8 + i%8][nt*16 + lr], i = 0..15 (32B per lane).
__global__ void moe_cvtW(const float* __restrict__ W, bf16* __restrict__ Ws,
                         int Kdim, int Ndim, long total) {
  const int KS = Kdim / 32, NT = Ndim / 16;
  for (long idx = (long)blockIdx.x * blockDim.x + threadIdx.x; idx < total;
       idx += (long)gridDim.x * blockDim.x) {
    int within = (int)(idx & 511);
    long frag  = idx >> 9;
    int i    = within & 15;
    int lane = within >> 4;
    int ks   = (int)(frag % KS);
    int nt   = (int)((frag / KS) % NT);
    int e    = (int)(frag / ((long)KS * NT));
    int k = ks * 32 + (i >> 3) * 16 + (lane >> 4) * 8 + (i & 7);
    int n = nt * 16 + (lane & 15);
    Ws[idx] = f2bf(W[(size_t)e * Kdim * Ndim + (size_t)k * Ndim + n]);
  }
}

// ---------------------------------------------------------------- gating
__global__ void moe_gate(const float* __restrict__ x, const float* __restrict__ Wg,
                         int* __restrict__ counts, int* __restrict__ tok,
                         float* __restrict__ wgt) {
  int t = blockIdx.x * blockDim.x + threadIdx.x;
  if (t >= S_TOK) return;

  float acc[NEXP];
#pragma unroll
  for (int e = 0; e < NEXP; e++) acc[e] = 0.0f;

  const float* xr = x + (size_t)t * DMODEL;
  for (int d = 0; d < DMODEL; d++) {
    float xv = xr[d];
#pragma unroll
    for (int e = 0; e < NEXP; e++) acc[e] = fmaf(xv, Wg[d * NEXP + e], acc[e]);
  }

  float mx = acc[0];
#pragma unroll
  for (int e = 1; e < NEXP; e++) mx = fmaxf(mx, acc[e]);
  float p[NEXP];
#pragma unroll
  for (int e = 0; e < NEXP; e++) p[e] = __expf(acc[e] - mx);
  int i1 = 0;
#pragma unroll
  for (int e = 1; e < NEXP; e++) if (p[e] > p[i1]) i1 = e;
  int i2 = (i1 == 0) ? 1 : 0;
#pragma unroll
  for (int e = 0; e < NEXP; e++) if (e != i1 && p[e] > p[i2]) i2 = e;

  float v1 = p[i1], v2 = p[i2];
  float inv = 1.0f / (v1 + v2);

  int s1 = atomicAdd(&counts[i1], 1);
  tok[i1 * S_TOK + s1] = t; wgt[i1 * S_TOK + s1] = v1 * inv;
  int s2 = atomicAdd(&counts[i2], 1);
  tok[i2 * S_TOK + s2] = t; wgt[i2 * S_TOK + s2] = v2 * inv;
}

// ---------------------------------------------------------------- expert FFN
// grid = (NEXP, S_TOK/MTILE), block = 256 (8 wave32s). One block = 32 tokens
// of one expert. Two 16-row A fragments share each pre-swizzled B fragment.
__global__ __launch_bounds__(256)
void moe_ffn(const float* __restrict__ x,
             const bf16* __restrict__ W1s, const float* __restrict__ b1,
             const bf16* __restrict__ W2s, const float* __restrict__ b2,
             const int* __restrict__ counts, const int* __restrict__ tok,
             const float* __restrict__ wgt, float* __restrict__ out) {
  const int e    = blockIdx.x;
  const int cnt  = counts[e];
  const int base = blockIdx.y * MTILE;
  if (base >= cnt) return;   // uniform early-out, no barriers crossed

  extern __shared__ char smem[];
  bf16* As = (bf16*)smem;            // [MTILE][LDA]
  bf16* Hs = As + MTILE * LDA;       // [MTILE][LDH]
  __shared__ int   s_tok[MTILE];
  __shared__ float s_w[MTILE];

  const int tid = threadIdx.x;
  if (tid < MTILE) {
    int r = base + tid;
    bool v = r < cnt;
    s_tok[tid] = v ? tok[e * S_TOK + r] : 0;
    s_w[tid]   = v ? wgt[e * S_TOK + r] : 0.0f;   // pad rows -> weight 0
  }
  __syncthreads();

  // stage MTILE x DMODEL token rows: fp32 global (float4) -> bf16 LDS
  for (int i = tid; i < MTILE * (DMODEL / 4); i += 256) {
    int r = i / (DMODEL / 4), c = (i - r * (DMODEL / 4)) * 4;
    float4 v = make_float4(0.f, 0.f, 0.f, 0.f);
    if ((base + r) < cnt)
      v = *(const float4*)&x[(size_t)s_tok[r] * DMODEL + c];
    v4bf o; o[0] = f2bf(v.x); o[1] = f2bf(v.y); o[2] = f2bf(v.z); o[3] = f2bf(v.w);
    *(v4bf*)&As[r * LDA + c] = o;
  }
  __syncthreads();

  const int lane = tid & 31;
  const int wave = tid >> 5;
  const int half = lane >> 4;
  const int lr   = lane & 15;

  const float* b1e = b1 + (size_t)e * DFF;

  // ---- GEMM1: H[32,DFF] = relu(A[32,DM] @ W1 + b1) ----
  for (int nt = wave; nt < NT1; nt += 8) {
    const int ncol = nt * 16 + lr;
    const bf16* bptr = W1s + (((size_t)(e * NT1 + nt) * KS1) * 32 + lane) * 16;
    v8f acc0 = {}, acc1 = {};
    for (int ks = 0; ks < KS1; ks++) {
      const int kb = ks * 32;
      // A fragments: two aligned 16B runs per lane per ISA layout
      v8bf a0l = *(const v8bf*)&As[lr * LDA + kb + half * 8];
      v8bf a0h = *(const v8bf*)&As[lr * LDA + kb + 16 + half * 8];
      v8bf a1l = *(const v8bf*)&As[(16 + lr) * LDA + kb + half * 8];
      v8bf a1h = *(const v8bf*)&As[(16 + lr) * LDA + kb + 16 + half * 8];
      v16bf a0 = __builtin_shufflevector(a0l, a0h, 0,1,2,3,4,5,6,7,8,9,10,11,12,13,14,15);
      v16bf a1 = __builtin_shufflevector(a1l, a1h, 0,1,2,3,4,5,6,7,8,9,10,11,12,13,14,15);
      // B fragment: 32 contiguous bytes per lane (pre-swizzled bf16)
      v16bf b = *(const v16bf*)(bptr + (size_t)ks * 512);
      __builtin_prefetch(bptr + (size_t)(ks + 1) * 512, 0, 1);
      acc0 = __builtin_amdgcn_wmma_f32_16x16x32_bf16(false, a0, false, b, (short)0, acc0, false, false);
      acc1 = __builtin_amdgcn_wmma_f32_16x16x32_bf16(false, a1, false, b, (short)0, acc1, false, false);
    }
    float bias = b1e[ncol];
#pragma unroll
    for (int r = 0; r < 8; r++) {        // C layout: vgpr r -> M = r + 8*half
      int m = r + 8 * half;
      Hs[m * LDH + ncol]        = f2bf(fmaxf(acc0[r] + bias, 0.0f));
      Hs[(16 + m) * LDH + ncol] = f2bf(fmaxf(acc1[r] + bias, 0.0f));
    }
  }
  __syncthreads();

  const float* b2e = b2 + (size_t)e * DMODEL;

  // ---- GEMM2 + weighted combine: out += w * (H @ W2 + b2) ----
  for (int nt = wave; nt < NT2; nt += 8) {
    const int ncol = nt * 16 + lr;
    const bf16* bptr = W2s + (((size_t)(e * NT2 + nt) * KS2) * 32 + lane) * 16;
    v8f acc0 = {}, acc1 = {};
    for (int ks = 0; ks < KS2; ks++) {
      const int kb = ks * 32;
      v8bf a0l = *(const v8bf*)&Hs[lr * LDH + kb + half * 8];
      v8bf a0h = *(const v8bf*)&Hs[lr * LDH + kb + 16 + half * 8];
      v8bf a1l = *(const v8bf*)&Hs[(16 + lr) * LDH + kb + half * 8];
      v8bf a1h = *(const v8bf*)&Hs[(16 + lr) * LDH + kb + 16 + half * 8];
      v16bf a0 = __builtin_shufflevector(a0l, a0h, 0,1,2,3,4,5,6,7,8,9,10,11,12,13,14,15);
      v16bf a1 = __builtin_shufflevector(a1l, a1h, 0,1,2,3,4,5,6,7,8,9,10,11,12,13,14,15);
      v16bf b = *(const v16bf*)(bptr + (size_t)ks * 512);
      __builtin_prefetch(bptr + (size_t)(ks + 1) * 512, 0, 1);
      acc0 = __builtin_amdgcn_wmma_f32_16x16x32_bf16(false, a0, false, b, (short)0, acc0, false, false);
      acc1 = __builtin_amdgcn_wmma_f32_16x16x32_bf16(false, a1, false, b, (short)0, acc1, false, false);
    }
    float bias = b2e[ncol];
#pragma unroll
    for (int r = 0; r < 8; r++) {
      int m = r + 8 * half;
      float w0 = s_w[m], w1 = s_w[16 + m];
      if (w0 != 0.0f)
        atomicAdd(&out[(size_t)s_tok[m] * DMODEL + ncol], w0 * (acc0[r] + bias));
      if (w1 != 0.0f)
        atomicAdd(&out[(size_t)s_tok[16 + m] * DMODEL + ncol], w1 * (acc1[r] + bias));
    }
  }
}

// ---------------------------------------------------------------- launch
extern "C" void kernel_launch(void* const* d_in, const int* in_sizes, int n_in,
                              void* d_out, int out_size, void* d_ws, size_t ws_size,
                              hipStream_t stream) {
  const float* x  = (const float*)d_in[0];
  const float* Wg = (const float*)d_in[1];
  const float* W1 = (const float*)d_in[2];
  const float* b1 = (const float*)d_in[3];
  const float* W2 = (const float*)d_in[4];
  const float* b2 = (const float*)d_in[5];
  float* out = (float*)d_out;

  // ws: counts[64] | tok[E*S] | wgt[E*S] | W1s bf16 | W2s bf16  (~76 MB)
  char* w = (char*)d_ws;
  int*   counts = (int*)w;                 w += 256;
  int*   tok    = (int*)w;                 w += sizeof(int)   * NEXP * S_TOK;
  float* wgt    = (float*)w;               w += sizeof(float) * NEXP * S_TOK;
  bf16*  W1s    = (bf16*)w;                w += sizeof(bf16)  * W1S_ELEMS;
  bf16*  W2s    = (bf16*)w;

  moe_zero<<<256, 256, 0, stream>>>(out, S_TOK * DMODEL, counts);
  moe_gate<<<S_TOK / 256, 256, 0, stream>>>(x, Wg, counts, tok, wgt);
  moe_cvtW<<<4096, 256, 0, stream>>>(W1, W1s, DMODEL, DFF, (long)W1S_ELEMS);
  moe_cvtW<<<4096, 256, 0, stream>>>(W2, W2s, DFF, DMODEL, (long)W2S_ELEMS);

  size_t lds_bytes = (size_t)(MTILE * LDA + MTILE * LDH) * sizeof(bf16);  // ~241 KB
  dim3 grid(NEXP, S_TOK / MTILE);
  moe_ffn<<<grid, 256, lds_bytes, stream>>>(x, W1s, b1, W2s, b2, counts, tok, wgt, out);
}
